// SlateRecommender_38611755991272
// MI455X (gfx1250) — matrix operand order repaired and checked by reading, not compiled, not run on hardware
//
#include <hip/hip_runtime.h>
#include <hip/hip_bf16.h>
#include <math.h>

typedef __attribute__((ext_vector_type(16))) _Float16 v16h;
typedef __attribute__((ext_vector_type(8)))  _Float16 v8h;
typedef __attribute__((ext_vector_type(8)))  float    v8f;
typedef __attribute__((ext_vector_type(4)))  unsigned int u32x4;
typedef __attribute__((ext_vector_type(8)))  int i32x8;
typedef __attribute__((ext_vector_type(4)))  int i32x4;

#define B_      64
#define S_      200
#define EMB_    64
#define STATE_  256
#define HID_    512
#define SLATE_  10
#define HEADS_  4
#define HD_     16
#define NITEMS_ 10000

// ---------------------------------------------------------------------------
// Fast tanh via hardware v_exp_f32
// ---------------------------------------------------------------------------
__device__ __forceinline__ float fast_tanhf(float x) {
  float t = __expf(-2.0f * fabsf(x));
  float r = (1.0f - t) / (1.0f + t);
  return copysignf(r, x);
}

// ---------------------------------------------------------------------------
// TDM: 1-D contiguous copy global(f16) -> LDS via tensor_load_to_lds.
// ---------------------------------------------------------------------------
__device__ __forceinline__ void tdm_copy_1d(unsigned lds_addr, const void* gptr,
                                            unsigned tile_elems, unsigned tensor_elems) {
  unsigned long long ga = (unsigned long long)gptr;
  u32x4 g0 = {};
  g0[0] = 1u;                                   // [1:0] count=1
  g0[1] = lds_addr;                             // [63:32] lds_addr (bytes)
  g0[2] = (unsigned)(ga & 0xffffffffu);         // [95:64] global_addr lo
  g0[3] = (unsigned)((ga >> 32) & 0x1ffffffu)   // [120:96] global_addr hi
        | (2u << 30);                           // [127:126] type=2
  i32x8 g1 = {};
  g1[0] = (int)(1u << 16);                      // [17:16] data_size=1 -> 2 bytes
  g1[1] = (int)((tensor_elems & 0xffffu) << 16);        // tensor_dim0[15:0]
  g1[2] = (int)(((tensor_elems >> 16) & 0xffffu)        // tensor_dim0[31:16]
        | (1u << 16));                                  // tensor_dim1 = 1
  g1[3] = (int)((tile_elems & 0xffffu) << 16);          // tile_dim0
  g1[4] = 1;                                            // tile_dim1 = 1
  g1[5] = (int)tensor_elems;                            // tensor_dim0_stride lo32
  i32x4 z4 = {};
#if defined(__clang_major__) && (__clang_major__ >= 23)
  i32x8 z8 = {};
  __builtin_amdgcn_tensor_load_to_lds(g0, g1, z4, z4, z8, 0);
#else
  __builtin_amdgcn_tensor_load_to_lds(g0, g1, z4, z4, 0);
#endif
}

__device__ __forceinline__ unsigned lds_off(const void* p) {
  return (unsigned)(unsigned long long)p;   // generic LDS ptr: low 32 bits = LDS offset
}

// A fragment (16-bit A 16x32, contiguous k = 16*(lane/16) + s, as B): two contiguous 8-half chunks per lane.
__device__ __forceinline__ v16h load_a_frag(const _Float16* lsA, int arow, int k0,
                                            int ab0, int ab1) {
  const _Float16* ap = lsA + arow * 64 + k0;
  v8h lo = *(const v8h*)(ap + ab0);
  v8h hi = *(const v8h*)(ap + ab1);
  return __builtin_shufflevector(lo, hi, 0, 1, 2, 3, 4, 5, 6, 7,
                                 8, 9, 10, 11, 12, 13, 14, 15);
}

// ---------------------------------------------------------------------------
// keys = item_emb[10000,64] @ key_w[64,512] + key_b  (WMMA f16 -> f32 acc)
// ---------------------------------------------------------------------------
__global__ __launch_bounds__(256)
void keys_gemm_wmma(const _Float16* __restrict__ itemH,   // [10000*64]
                    const _Float16* __restrict__ keyWT,   // [512*64], n-major
                    const float* __restrict__ key_b,
                    float* __restrict__ keys) {
  __shared__ __align__(32) _Float16 lsA[128 * 64];
  __shared__ __align__(32) _Float16 lsBT[64 * 64];
  const int tid = threadIdx.x, lane = tid & 31, w = tid >> 5;
  const int m0 = blockIdx.x * 128, n0 = blockIdx.y * 64;

  if (tid == 0) {
    tdm_copy_1d(lds_off(lsA),  itemH + m0 * 64, 128 * 64, NITEMS_ * 64);
    tdm_copy_1d(lds_off(lsBT), keyWT + n0 * 64,  64 * 64, HID_ * 64);
  }
  __builtin_amdgcn_s_wait_tensorcnt(0);
  __syncthreads();

  v8f acc[4] = {};
  const int arow = w * 16 + (lane & 15);
  const int ab0  = (lane < 16) ? 0 : 16;
  const int ab1  = (lane < 16) ? 8 : 24;
  const int bb   = (lane < 16) ? 0 : 16;
  const int bn   = lane & 15;

#pragma unroll
  for (int k0 = 0; k0 < 64; k0 += 32) {
    v16h a = load_a_frag(lsA, arow, k0, ab0, ab1);
#pragma unroll
    for (int nt = 0; nt < 4; ++nt) {
      v16h bfrag = *(const v16h*)&lsBT[(nt * 16 + bn) * 64 + k0 + bb];
      acc[nt] = __builtin_amdgcn_wmma_f32_16x16x32_f16(
          false, a, false, bfrag, (short)0, acc[nt], false, false);
    }
  }

  const int mbase = m0 + w * 16 + ((lane >= 16) ? 8 : 0);
  if (m0 + 128 <= NITEMS_) {
    // Full tile: branch-free epilogue (all blocks except the last M-block)
#pragma unroll
    for (int nt = 0; nt < 4; ++nt) {
      int c = n0 + nt * 16 + bn;
      float bias = key_b[c];
#pragma unroll
      for (int r = 0; r < 8; ++r) keys[(mbase + r) * HID_ + c] = acc[nt][r] + bias;
    }
  } else {
#pragma unroll
    for (int nt = 0; nt < 4; ++nt) {
      int c = n0 + nt * 16 + bn;
      float bias = key_b[c];
#pragma unroll
      for (int r = 0; r < 8; ++r) {
        int m = mbase + r;
        if (m < NITEMS_) keys[m * HID_ + c] = acc[nt][r] + bias;
      }
    }
  }
}

// ---------------------------------------------------------------------------
// QKV projection via WMMA: out[12800,64] = embH @ w[64,64] + bias
// (12800 = 100 * 128, so the epilogue is always full-tile.)
// ---------------------------------------------------------------------------
__global__ __launch_bounds__(256)
void qkv_gemm_wmma(const _Float16* __restrict__ embH,   // [12800*64]
                   const float* __restrict__ wmat,      // [64*64] (k-major)
                   const float* __restrict__ bias,
                   float* __restrict__ out) {
  __shared__ __align__(32) _Float16 lsA[128 * 64];
  __shared__ __align__(32) _Float16 lsBT[64 * 64];
  const int tid = threadIdx.x, lane = tid & 31, w = tid >> 5;
  const int m0 = blockIdx.x * 128;

  if (tid == 0)
    tdm_copy_1d(lds_off(lsA), embH + m0 * 64, 128 * 64, B_ * S_ * 64);
  for (int idx = tid; idx < 64 * 64; idx += 256) {
    int n = idx >> 6, k = idx & 63;
    lsBT[idx] = (_Float16)wmat[k * 64 + n];   // transpose + convert in-kernel
  }
  __builtin_amdgcn_s_wait_tensorcnt(0);
  __syncthreads();

  v8f acc[4] = {};
  const int arow = w * 16 + (lane & 15);
  const int ab0  = (lane < 16) ? 0 : 16;
  const int ab1  = (lane < 16) ? 8 : 24;
  const int bb   = (lane < 16) ? 0 : 16;
  const int bn   = lane & 15;

#pragma unroll
  for (int k0 = 0; k0 < 64; k0 += 32) {
    v16h a = load_a_frag(lsA, arow, k0, ab0, ab1);
#pragma unroll
    for (int nt = 0; nt < 4; ++nt) {
      v16h bfrag = *(const v16h*)&lsBT[(nt * 16 + bn) * 64 + k0 + bb];
      acc[nt] = __builtin_amdgcn_wmma_f32_16x16x32_f16(
          false, a, false, bfrag, (short)0, acc[nt], false, false);
    }
  }

  const int mbase = m0 + w * 16 + ((lane >= 16) ? 8 : 0);
#pragma unroll
  for (int nt = 0; nt < 4; ++nt) {
    int c = nt * 16 + bn;
    float bv = bias[c];
#pragma unroll
    for (int r = 0; r < 8; ++r) out[(mbase + r) * EMB_ + c] = acc[nt][r] + bv;
  }
}

// ---------------------------------------------------------------------------
// Pre-convert: item_emb -> f16, key_w -> transposed f16, gathered history -> f16
// ---------------------------------------------------------------------------
__global__ void convert_f16(const float* __restrict__ item_emb,
                            const float* __restrict__ key_w,
                            const int* __restrict__ hist,
                            const float* __restrict__ enc_emb,
                            _Float16* __restrict__ itemH,
                            _Float16* __restrict__ keyWT,
                            _Float16* __restrict__ embH) {
  int idx = blockIdx.x * blockDim.x + threadIdx.x;
  if (idx < NITEMS_ * EMB_) itemH[idx] = (_Float16)item_emb[idx];
  if (idx < HID_ * EMB_) {
    int n = idx >> 6, k = idx & 63;
    keyWT[idx] = (_Float16)key_w[k * HID_ + n];
  }
  if (idx < B_ * S_ * EMB_) {
    int e = idx & 63, bs = idx >> 6;
    embH[idx] = (_Float16)enc_emb[hist[bs] * EMB_ + e];
  }
}

// ---------------------------------------------------------------------------
// Encoder attention / pooling / FC+LN (VALU; negligible runtime share)
// ---------------------------------------------------------------------------
__global__ __launch_bounds__(256)
void mha_attn(const float* __restrict__ Qb, const float* __restrict__ Kb,
              const float* __restrict__ Vb, const unsigned char* __restrict__ mask,
              float* __restrict__ Ob) {
  __shared__ float Ks[S_ * HD_];
  __shared__ float Vs[S_ * HD_];
  __shared__ unsigned char Ms[S_];
  int b = blockIdx.x, h = blockIdx.y, tid = threadIdx.x;
  for (int idx = tid; idx < S_ * HD_; idx += 256) {
    int s = idx >> 4, d = idx & 15;
    int g = (b * S_ + s) * EMB_ + h * HD_ + d;
    Ks[idx] = Kb[g]; Vs[idx] = Vb[g];
  }
  for (int s = tid; s < S_; s += 256) Ms[s] = mask[b * S_ + s];
  __syncthreads();
  if (tid >= S_) return;
  float q[HD_], o[HD_];
#pragma unroll
  for (int d = 0; d < HD_; ++d) { q[d] = Qb[(b * S_ + tid) * EMB_ + h * HD_ + d]; o[d] = 0.0f; }
  float m = -INFINITY, ssum = 0.0f;
  for (int k = 0; k < S_; ++k) {
    if (Ms[k]) continue;
    float sc = 0.0f;
#pragma unroll
    for (int d = 0; d < HD_; ++d) sc += q[d] * Ks[k * HD_ + d];
    sc *= 0.25f;  // 1/sqrt(16)
    if (sc > m) {
      float r = __expf(m - sc);
      ssum = ssum * r + 1.0f;
#pragma unroll
      for (int d = 0; d < HD_; ++d) o[d] = o[d] * r + Vs[k * HD_ + d];
      m = sc;
    } else {
      float e = __expf(sc - m);
      ssum += e;
#pragma unroll
      for (int d = 0; d < HD_; ++d) o[d] += e * Vs[k * HD_ + d];
    }
  }
  float inv = 1.0f / ssum;
#pragma unroll
  for (int d = 0; d < HD_; ++d) Ob[(b * S_ + tid) * EMB_ + h * HD_ + d] = o[d] * inv;
}

__global__ void mean_pool(const float* __restrict__ Ob,
                          const unsigned char* __restrict__ mask,
                          float* __restrict__ meanO) {
  int idx = blockIdx.x * blockDim.x + threadIdx.x;
  if (idx >= B_ * EMB_) return;
  int b = idx >> 6, e = idx & 63;
  float s = 0.0f, cnt = 0.0f;
  for (int k = 0; k < S_; ++k) {
    if (!mask[b * S_ + k]) { s += Ob[(b * S_ + k) * EMB_ + e]; cnt += 1.0f; }
  }
  meanO[idx] = s / fmaxf(cnt, 1.0f);
}

__global__ __launch_bounds__(256)
void fc_ln(const float* __restrict__ meanO, const float* __restrict__ wo,
           const float* __restrict__ bo, const float* __restrict__ fc_w,
           const float* __restrict__ fc_b, const float* __restrict__ ln_g,
           const float* __restrict__ ln_b, float* __restrict__ state) {
  __shared__ float pool[EMB_];
  __shared__ float red[256];
  int b = blockIdx.x, tid = threadIdx.x;
  if (tid < EMB_) {
    float acc = bo[tid];
#pragma unroll 8
    for (int i = 0; i < EMB_; ++i) acc += meanO[b * EMB_ + i] * wo[i * EMB_ + tid];
    pool[tid] = acc;
  }
  __syncthreads();
  float h = fc_b[tid];
#pragma unroll 8
  for (int i = 0; i < EMB_; ++i) h += pool[i] * fc_w[i * STATE_ + tid];
  h = fmaxf(h, 0.0f);
  red[tid] = h; __syncthreads();
  for (int s = 128; s > 0; s >>= 1) { if (tid < s) red[tid] += red[tid + s]; __syncthreads(); }
  float mu = red[0] / 256.0f;
  __syncthreads();
  float d = h - mu;
  red[tid] = d * d; __syncthreads();
  for (int s = 128; s > 0; s >>= 1) { if (tid < s) red[tid] += red[tid + s]; __syncthreads(); }
  float var = red[0] / 256.0f;
  state[b * STATE_ + tid] = d * rsqrtf(var + 1e-5f) * ln_g[tid] + ln_b[tid];
}

// ---------------------------------------------------------------------------
// Slate loop kernels
// ---------------------------------------------------------------------------
__global__ void qpos_kernel(const float* __restrict__ state, const float* __restrict__ ctx,
                            const float* __restrict__ pos_emb, const float* __restrict__ query_w,
                            const float* __restrict__ query_b, int pos,
                            float* __restrict__ qbuf) {
  int idx = blockIdx.x * blockDim.x + threadIdx.x;
  if (idx >= B_ * HID_) return;
  int b = idx >> 9, o = idx & (HID_ - 1);
  float acc = query_b[o];
  const float* st = state + b * STATE_;
#pragma unroll 4
  for (int i = 0; i < STATE_; ++i) acc += st[i] * query_w[i * HID_ + o];
  const float* cx = ctx + b * EMB_;
  const float* pe = pos_emb + pos * EMB_;
#pragma unroll 4
  for (int i = 0; i < EMB_; ++i) acc += (cx[i] + pe[i]) * query_w[(STATE_ + i) * HID_ + o];
  qbuf[idx] = acc;
}

__global__ __launch_bounds__(256)
void scores_kernel(const float* __restrict__ qbuf, const float* __restrict__ keys,
                   const float* __restrict__ attn_w, const float* __restrict__ attn_b,
                   float* __restrict__ scores) {
  __shared__ float qs[HID_];
  __shared__ float aw[HID_];
  int b = blockIdx.y, tid = threadIdx.x;
  for (int i = tid; i < HID_; i += 256) { qs[i] = qbuf[b * HID_ + i]; aw[i] = attn_w[i]; }
  __syncthreads();
  int n = blockIdx.x * 256 + tid;
  if (n >= NITEMS_) return;
  const float* kr = keys + n * HID_;
  float acc = attn_b[0];
#pragma unroll 4
  for (int h = 0; h < HID_; ++h) acc += fast_tanhf(qs[h] + kr[h]) * aw[h];
  scores[b * NITEMS_ + n] = acc;
}

__global__ __launch_bounds__(256)
void sample_kernel(const float* __restrict__ scores, unsigned char* __restrict__ sel,
                   const float* __restrict__ item_emb, float* __restrict__ ctx,
                   float* __restrict__ out, int pos) {
  __shared__ float s_m[256], s_s[256], s_bv[256];
  __shared__ int s_bi[256];
  __shared__ int s_act;
  int b = blockIdx.x, tid = threadIdx.x;
  float m = -INFINITY, ssum = 0.0f, bv = -INFINITY;
  int bi = 0;
  for (int n = tid; n < NITEMS_; n += 256) {
    float logit = sel[b * NITEMS_ + n] ? -INFINITY : scores[b * NITEMS_ + n];
    if (logit != -INFINITY) {
      if (logit > m) { ssum = ssum * __expf(m - logit) + 1.0f; m = logit; }
      else           { ssum += __expf(logit - m); }
    }
    unsigned x = (unsigned)b * 0x9E3779B1u ^ (unsigned)n * 0x85EBCA77u ^ (unsigned)pos * 0xC2B2AE3Du;
    x ^= x >> 16; x *= 0x7FEB352Du; x ^= x >> 15; x *= 0x846CA68Bu; x ^= x >> 16;
    float u = ((float)(x >> 8) + 0.5f) * (1.0f / 16777216.0f);
    float g = -__logf(-__logf(u));
    float pv = logit + g;
    if (pv > bv) { bv = pv; bi = n; }
  }
  s_m[tid] = m; s_s[tid] = ssum; s_bv[tid] = bv; s_bi[tid] = bi;
  __syncthreads();
  for (int str = 128; str > 0; str >>= 1) {
    if (tid < str) {
      float m2 = s_m[tid + str], s2 = s_s[tid + str];
      if (m2 != -INFINITY) {
        if (s_m[tid] == -INFINITY)   { s_m[tid] = m2; s_s[tid] = s2; }
        else if (m2 > s_m[tid])      { s_s[tid] = s_s[tid] * __expf(s_m[tid] - m2) + s2; s_m[tid] = m2; }
        else                         { s_s[tid] += s2 * __expf(m2 - s_m[tid]); }
      }
      if (s_bv[tid + str] > s_bv[tid]) { s_bv[tid] = s_bv[tid + str]; s_bi[tid] = s_bi[tid + str]; }
    }
    __syncthreads();
  }
  if (tid == 0) {
    int action = s_bi[0];
    float logZ = s_m[0] + __logf(s_s[0]);
    out[b * SLATE_ + pos] = (float)action;
    out[B_ * SLATE_ + b * SLATE_ + pos] = scores[b * NITEMS_ + action] - logZ;
    sel[b * NITEMS_ + action] = 1;
    s_act = action;
  }
  __syncthreads();
  if (tid < EMB_) ctx[b * EMB_ + tid] = item_emb[s_act * EMB_ + tid];
}

// ---------------------------------------------------------------------------
// Host launch
// ---------------------------------------------------------------------------
extern "C" void kernel_launch(void* const* d_in, const int* in_sizes, int n_in,
                              void* d_out, int out_size, void* d_ws, size_t ws_size,
                              hipStream_t stream) {
  (void)in_sizes; (void)n_in; (void)out_size; (void)ws_size;
  const int*           history  = (const int*)d_in[0];
  const unsigned char* hmask    = (const unsigned char*)d_in[1];
  const float* enc_emb  = (const float*)d_in[2];
  const float* wq = (const float*)d_in[3];  const float* bq = (const float*)d_in[4];
  const float* wk = (const float*)d_in[5];  const float* bk = (const float*)d_in[6];
  const float* wv = (const float*)d_in[7];  const float* bv = (const float*)d_in[8];
  const float* wo = (const float*)d_in[9];  const float* bo = (const float*)d_in[10];
  const float* fc_w = (const float*)d_in[11]; const float* fc_b = (const float*)d_in[12];
  const float* ln_g = (const float*)d_in[13]; const float* ln_b = (const float*)d_in[14];
  const float* item_emb = (const float*)d_in[15];
  const float* pos_emb  = (const float*)d_in[16];
  const float* query_w  = (const float*)d_in[17]; const float* query_b = (const float*)d_in[18];
  const float* key_w    = (const float*)d_in[19]; const float* key_b   = (const float*)d_in[20];
  const float* attn_w   = (const float*)d_in[21]; const float* attn_b  = (const float*)d_in[22];

  // Workspace layout
  float* keys   = (float*)d_ws;               // 10000*512
  float* Qb     = keys   + NITEMS_ * HID_;    // 64*200*64
  float* Kb     = Qb     + B_ * S_ * EMB_;
  float* Vb     = Kb     + B_ * S_ * EMB_;
  float* Ob     = Vb     + B_ * S_ * EMB_;
  float* meanO  = Ob     + B_ * S_ * EMB_;    // 64*64
  float* state  = meanO  + B_ * EMB_;         // 64*256
  float* ctx    = state  + B_ * STATE_;       // 64*64
  float* qbuf   = ctx    + B_ * EMB_;         // 64*512
  float* scores = qbuf   + B_ * HID_;         // 64*10000
  _Float16* itemH = (_Float16*)(scores + B_ * NITEMS_);  // 10000*64
  _Float16* keyWT = itemH + NITEMS_ * EMB_;              // 512*64 (transposed)
  _Float16* embH  = keyWT + HID_ * EMB_;                 // 64*200*64
  unsigned char* sel = (unsigned char*)(embH + B_ * S_ * EMB_);  // 64*10000 bytes

  (void)hipMemsetAsync(sel, 0, (size_t)B_ * NITEMS_, stream);
  (void)hipMemsetAsync(ctx, 0, (size_t)B_ * EMB_ * sizeof(float), stream);

  // Pre-convert to f16 (item table, key_w^T, gathered history embeddings)
  convert_f16<<<(B_ * S_ * EMB_ + 255) / 256, 256, 0, stream>>>(
      item_emb, key_w, history, enc_emb, itemH, keyWT, embH);

  // keys GEMM (WMMA + TDM staging)
  keys_gemm_wmma<<<dim3((NITEMS_ + 127) / 128, HID_ / 64), 256, 0, stream>>>(
      itemH, keyWT, key_b, keys);

  // Encoder: QKV projections (WMMA), attention, pooling, FC+LN
  qkv_gemm_wmma<<<(B_ * S_) / 128, 256, 0, stream>>>(embH, wq, bq, Qb);
  qkv_gemm_wmma<<<(B_ * S_) / 128, 256, 0, stream>>>(embH, wk, bk, Kb);
  qkv_gemm_wmma<<<(B_ * S_) / 128, 256, 0, stream>>>(embH, wv, bv, Vb);
  mha_attn<<<dim3(B_, HEADS_), 256, 0, stream>>>(Qb, Kb, Vb, hmask, Ob);
  mean_pool<<<(B_ * EMB_ + 255) / 256, 256, 0, stream>>>(Ob, hmask, meanO);
  fc_ln<<<B_, 256, 0, stream>>>(meanO, wo, bo, fc_w, fc_b, ln_g, ln_b, state);

  // Autoregressive slate loop
  for (int pos = 0; pos < SLATE_; ++pos) {
    qpos_kernel<<<(B_ * HID_ + 255) / 256, 256, 0, stream>>>(
        state, ctx, pos_emb, query_w, query_b, pos, qbuf);
    scores_kernel<<<dim3((NITEMS_ + 255) / 256, B_), 256, 0, stream>>>(
        qbuf, keys, attn_w, attn_b, scores);
    sample_kernel<<<B_, 256, 0, stream>>>(scores, sel, item_emb, ctx, (float*)d_out, pos);
  }
}
